// Linear_20847771255232
// MI455X (gfx1250) — matrix operand order repaired and checked by reading, not compiled, not run on hardware
//
#include <hip/hip_runtime.h>
#include <hip/hip_bf16.h>

// ---------------------------------------------------------------------------
// y = x @ (W_base + S * W_add)^T + b, where the block-adapter collapses to
// W_add[o*64+e, i*64+d] = U[d,i,o] * V[o,d,e].
//   1) prep_w : fold adapter into effective weight, convert -> bf16
//   2) prep_x : x -> bf16
//   3) gemm   : bf16 WMMA (v_wmma_f32_16x16x32_bf16), fp32 acc, fused bias.
//               Double-buffered LDS fed by global_load_async_to_lds_b128
//               (ASYNCcnt pipeline); K-loop unrolled by 2 so the buffer index
//               is static -> straight-line steady state (no cndmask/branches).
// Workspace: [ W_bf16 : 32MB ][ x_bf16 : 32MB ]
// ---------------------------------------------------------------------------

typedef unsigned short u16;
typedef u16   u16x8   __attribute__((ext_vector_type(8)));
typedef u16   u16x16  __attribute__((ext_vector_type(16)));
typedef __bf16 bf16x16 __attribute__((ext_vector_type(16)));
typedef float f32x8   __attribute__((ext_vector_type(8)));

#define DIN   4096
#define DOUT  4096
#define MTOT  4096   // B*N = 2*2048
#define BM    128
#define BN    128
#define BK    32
#define LDS_K 40     // padded stride in halfwords (80B rows: 16B aligned, bank-spread)

static __device__ __forceinline__ u16 f32_to_bf16(float f) {
    unsigned u = __float_as_uint(f);
    u += 0x7FFFu + ((u >> 16) & 1u);   // round-to-nearest-even
    return (u16)(u >> 16);
}

static __device__ __forceinline__ bf16x16 ld_frag(const u16* p0, const u16* p1) {
    u16x8 c0 = *(const u16x8*)p0;
    u16x8 c1 = *(const u16x8*)p1;
    u16x16 v = __builtin_shufflevector(c0, c1, 0,1,2,3,4,5,6,7,8,9,10,11,12,13,14,15);
    return __builtin_bit_cast(bf16x16, v);
}

// Async DMA 16 bytes global -> LDS. VDST holds the LDS byte offset
// (flat shared addresses carry the LDS offset in addr[31:0], per ISA 10.2).
static __device__ __forceinline__ void async_ld16(unsigned lds_off, const void* g) {
    asm volatile("global_load_async_to_lds_b128 %0, %1, off"
                 :: "v"(lds_off), "v"((unsigned long long)(uintptr_t)g)
                 : "memory");
}
static __device__ __forceinline__ void wait_async4() {
#if __has_builtin(__builtin_amdgcn_s_wait_asynccnt)
    __builtin_amdgcn_s_wait_asynccnt(4);
#else
    asm volatile("s_wait_asynccnt 0x4" ::: "memory");
#endif
}
static __device__ __forceinline__ void wait_async0() {
#if __has_builtin(__builtin_amdgcn_s_wait_asynccnt)
    __builtin_amdgcn_s_wait_asynccnt(0);
#else
    asm volatile("s_wait_asynccnt 0x0" ::: "memory");
#endif
}

// ---------------------------------------------------------------------------
__global__ __launch_bounds__(256)
void prep_w(const float* __restrict__ Wb, const float* __restrict__ U,
            const float* __restrict__ V,  const float* __restrict__ S,
            u16* __restrict__ Wout)
{
    size_t idx = (size_t)blockIdx.x * 256 + threadIdx.x;
    int n = (int)(idx >> 12);
    int k = (int)(idx & 4095);
    int o = n >> 6, e = n & 63;
    int i = k >> 6, d = k & 63;
    float s = S[0];                                  // * SCALING (== 1.0f)
    float u = U[((size_t)(d << 6) + i) * 64 + o];    // U[d,i,o]
    float v = V[((size_t)(o << 6) + d) * 64 + e];    // V[o,d,e]
    Wout[idx] = f32_to_bf16(Wb[idx] + s * u * v);
}

__global__ __launch_bounds__(256)
void prep_x(const float* __restrict__ X, u16* __restrict__ Xout)
{
    size_t idx = (size_t)blockIdx.x * 256 + threadIdx.x;
    Xout[idx] = f32_to_bf16(X[idx]);
}

// ---------------------------------------------------------------------------
// C[m,n] = sum_k A[m,k]*B[n,k] + bias[n]
// 128x128 tile / workgroup, 8 waves (2x4), each wave 4x2 tiles of 16x16.
// ---------------------------------------------------------------------------
__global__ __launch_bounds__(256)
void gemm_bf16_wmma(const u16* __restrict__ A,   // [MTOT][DIN]  bf16 bits
                    const u16* __restrict__ Bw,  // [DOUT][DIN]  bf16 bits
                    const float* __restrict__ bias,
                    float* __restrict__ C)       // [MTOT][DOUT] fp32
{
    __shared__ __align__(16) u16 lA[2][BM * LDS_K];
    __shared__ __align__(16) u16 lB[2][BN * LDS_K];

    const int tid   = threadIdx.x;
    const int lane  = tid & 31;
    const int wave  = tid >> 5;
    const int waveM = wave >> 2;   // 0..1 -> 4 M-tiles each
    const int waveN = wave & 3;    // 0..3 -> 2 N-tiles each

    const int blkM = blockIdx.y * BM;
    const int blkN = blockIdx.x * BN;

    f32x8 acc[4][2];
#pragma unroll
    for (int i = 0; i < 4; ++i)
#pragma unroll
        for (int j = 0; j < 2; ++j)
            acc[i][j] = (f32x8)0.0f;

    // Tile fill: 128 rows x 64B = 512 x 16B DMAs; 4 per thread (2 A + 2 B).
    const int t0 = tid, t1 = tid + 256;
    const int r0 = t0 >> 2, q0 = t0 & 3;        // row 0..127, 16B quad 0..3
    const int r1 = t1 >> 2, q1 = t1 & 3;
    const u16* Ar0 = A  + (size_t)(blkM + r0) * DIN + q0 * 8;
    const u16* Ar1 = A  + (size_t)(blkM + r1) * DIN + q1 * 8;
    const u16* Br0 = Bw + (size_t)(blkN + r0) * DIN + q0 * 8;
    const u16* Br1 = Bw + (size_t)(blkN + r1) * DIN + q1 * 8;

    unsigned a0off[2], a1off[2], b0off[2], b1off[2];
#pragma unroll
    for (int b = 0; b < 2; ++b) {
        a0off[b] = (unsigned)(uintptr_t)&lA[b][r0 * LDS_K + q0 * 8];
        a1off[b] = (unsigned)(uintptr_t)&lA[b][r1 * LDS_K + q1 * 8];
        b0off[b] = (unsigned)(uintptr_t)&lB[b][r0 * LDS_K + q0 * 8];
        b1off[b] = (unsigned)(uintptr_t)&lB[b][r1 * LDS_K + q1 * 8];
    }

    const int l16   = lane & 15;
    const int halfk = lane >> 4;   // 0/1: K-chunk set for this half-wave

    // Issue one tile's 4 DMAs into buffer b (compile-time b at every call).
    auto ISSUE = [&](int koff, int b) {
        async_ld16(a0off[b], Ar0 + koff);
        async_ld16(a1off[b], Ar1 + koff);
        async_ld16(b0off[b], Br0 + koff);
        async_ld16(b1off[b], Br1 + koff);
    };

    // Consume buffer b: 12 ds_load_b128 -> 8 WMMA.
    auto COMPUTE = [&](int b) {
        const u16* baseA = &lA[b][0];
        const u16* baseB = &lB[b][0];
        // A fragment (ISA 16-bit A layout): lanes 0-15 hold K {0-7,16-23},
        // lanes 16-31 hold K {8-15,24-31}, row M = lane&15.
        bf16x16 af[4];
#pragma unroll
        for (int i = 0; i < 4; ++i) {
            const u16* p = baseA + ((waveM * 4 + i) * 16 + l16) * LDS_K + halfk * 8;
            af[i] = ld_frag(p, p + 16);
        }
        // B fragment: lane n holds column n, same K order as A
        // (lanes 0-15: K {0-7,16-23}, lanes 16-31: K {8-15,24-31}).
        bf16x16 bfr[2];
#pragma unroll
        for (int j = 0; j < 2; ++j) {
            const u16* p = baseB + ((waveN * 2 + j) * 16 + l16) * LDS_K + halfk * 8;
            bfr[j] = ld_frag(p, p + 16);
        }
#pragma unroll
        for (int i = 0; i < 4; ++i)
#pragma unroll
            for (int j = 0; j < 2; ++j)
                acc[i][j] = __builtin_amdgcn_wmma_f32_16x16x32_bf16(
                    false, af[i], false, bfr[j],
                    (short)0, acc[i][j], false, false);
    };

    // Prologue: DMA tile 0 into buffer 0.
    ISSUE(0, 0);

    // Steady state: unconditional issue/wait/compute, static buffer index.
    // Processes tiles kt (buf0) and kt+BK (buf1) per backedge.
    int kt = 0;
    for (; kt + 2 * BK < DIN; kt += 2 * BK) {
        ISSUE(kt + BK, 1);
        __builtin_prefetch(Ar0 + kt + 2 * BK, 0, 1);   // global_prefetch_b8
        __builtin_prefetch(Br0 + kt + 2 * BK, 0, 1);
        wait_async4();            // tile kt landed; tile kt+BK may be in flight
        __syncthreads();
        COMPUTE(0);
        __syncthreads();

        ISSUE(kt + 2 * BK, 0);
        __builtin_prefetch(Ar0 + kt + 3 * BK, 0, 1);
        __builtin_prefetch(Br0 + kt + 3 * BK, 0, 1);
        wait_async4();
        __syncthreads();
        COMPUTE(1);
        __syncthreads();
    }

    // Tail: tiles kt (buf0, already issued) and kt+BK (buf1).
    ISSUE(kt + BK, 1);
    wait_async4();
    __syncthreads();
    COMPUTE(0);
    __syncthreads();
    wait_async0();
    __syncthreads();
    COMPUTE(1);

    // Epilogue: C/D layout — VGPR r holds M = r + 8*(lane>=16), N = lane&15.
    const int halfm = lane >> 4;
#pragma unroll
    for (int j = 0; j < 2; ++j) {
        int gn = blkN + (waveN * 2 + j) * 16 + l16;
        float bj = bias[gn];
#pragma unroll
        for (int i = 0; i < 4; ++i) {
            int gm0 = blkM + (waveM * 4 + i) * 16 + halfm * 8;
#pragma unroll
            for (int r = 0; r < 8; ++r)
                C[(size_t)(gm0 + r) * DOUT + gn] = acc[i][j][r] + bj;
        }
    }
}

// ---------------------------------------------------------------------------
extern "C" void kernel_launch(void* const* d_in, const int* in_sizes, int n_in,
                              void* d_out, int out_size, void* d_ws, size_t ws_size,
                              hipStream_t stream)
{
    const float* x  = (const float*)d_in[0];   // [2,2048,4096]
    const float* Wb = (const float*)d_in[1];   // [4096,4096]
    const float* bb = (const float*)d_in[2];   // [4096]
    const float* U  = (const float*)d_in[3];   // [64,64,64]
    const float* V  = (const float*)d_in[4];   // [64,64,64]
    const float* S  = (const float*)d_in[5];   // [1]
    float* out = (float*)d_out;                // [2,2048,4096]

    u16* Wbf = (u16*)d_ws;                     // 32 MB
    u16* Xbf = Wbf + (size_t)DOUT * DIN;       // 32 MB

    prep_w<<<(DOUT * DIN) / 256, 256, 0, stream>>>(Wb, U, V, S, Wbf);
    prep_x<<<(MTOT * DIN) / 256, 256, 0, stream>>>(x, Xbf);

    dim3 grid(DOUT / BN, MTOT / BM);           // (32, 32)
    gemm_bf16_wmma<<<grid, 256, 0, stream>>>(Xbf, Wbf, bb, out);
}
